// MarkerGAT_31499290149073
// MI455X (gfx1250) — matrix operand order repaired and checked by reading, not compiled, not run on hardware
//
#include <hip/hip_runtime.h>

#define NN 4096      // nodes
#define EMBD 128
#define QKVD 384

typedef __bf16 bf16x16 __attribute__((ext_vector_type(16)));
typedef __bf16 bf16x8  __attribute__((ext_vector_type(8)));
typedef float  f32x8   __attribute__((ext_vector_type(8)));
typedef unsigned int u32x4 __attribute__((ext_vector_type(4)));
typedef int i32x4 __attribute__((ext_vector_type(4)));
typedef int i32x8 __attribute__((ext_vector_type(8)));

#if defined(__gfx1250__) && __has_builtin(__builtin_amdgcn_tensor_load_to_lds)
#define USE_TDM 1
#else
#define USE_TDM 0
#endif

union BFrag { bf16x16 v; bf16x8 h[2]; };

__device__ inline unsigned short f2bf(float f) {
    unsigned int u = __float_as_uint(f);
    unsigned int r = (u + 0x7FFFu + ((u >> 16) & 1u)) >> 16;
    return (unsigned short)r;
}

__device__ inline bf16x16 load_a_frag(const unsigned short* rowPtr, int k0) {
    // A-matrix 16x32 bf16 layout: this lane holds K = {k0..k0+15}
    BFrag u;
    u.h[0] = *reinterpret_cast<const bf16x8*>(rowPtr + k0);
    u.h[1] = *reinterpret_cast<const bf16x8*>(rowPtr + k0 + 8);
    return u.v;
}

__device__ inline f32x8 wmma_bf16(bf16x16 a, bf16x16 b, f32x8 c) {
    return __builtin_amdgcn_wmma_f32_16x16x32_bf16(false, a, false, b, (short)0, c,
                                                   false, false);
}

__device__ inline void atomicMaxF(float* a, float v) {
    int old = __float_as_int(*a);
    while (__int_as_float(old) < v) {
        int prev = atomicCAS((int*)a, old, __float_as_int(v));
        if (prev == old) break;
        old = prev;
    }
}

// ---------------------------------------------------------------- utilities
__global__ void k_fill(float* p, float v, int n) {
    int i = blockIdx.x * blockDim.x + threadIdx.x;
    if (i < n) p[i] = v;
}

__global__ void k_f2bf(const float* __restrict__ in, unsigned short* __restrict__ out, int n) {
    int i = blockIdx.x * blockDim.x + threadIdx.x;
    if (i < n) out[i] = f2bf(in[i]);
}

// ---------------------------------------------------------------- GAT layer 1
__global__ void k_gat1_node(const float* __restrict__ x, const float* __restrict__ W1,
                            float* __restrict__ h1) {
    int n = blockIdx.x, j = threadIdx.x;          // 128 threads
    float acc = 0.f;
#pragma unroll
    for (int c = 0; c < 6; ++c) acc += x[n * 6 + c] * W1[c * 128 + j];
    h1[n * 128 + j] = acc;
}

__global__ void k_gat1_coef(const float* __restrict__ h1, const float* __restrict__ as,
                            const float* __restrict__ ad, float* __restrict__ es,
                            float* __restrict__ ed) {
    int idx = blockIdx.x * blockDim.x + threadIdx.x;   // N*4
    if (idx >= NN * 4) return;
    int n = idx >> 2, h = idx & 3;
    float s = 0.f, d = 0.f;
#pragma unroll
    for (int f = 0; f < 32; ++f) {
        float v = h1[n * 128 + h * 32 + f];
        s += v * as[h * 32 + f];
        d += v * ad[h * 32 + f];
    }
    es[idx] = s;
    ed[idx] = d;
}

__global__ void k_edge1_max(const int* __restrict__ ei, const float* __restrict__ es,
                            const float* __restrict__ ed, float* __restrict__ mseg,
                            int E, int ET) {
    int e = blockIdx.x * blockDim.x + threadIdx.x;
    if (e >= ET) return;
    int s, d;
    if (e < E) { s = ei[e]; d = ei[E + e]; } else { s = d = e - E; }
#pragma unroll
    for (int h = 0; h < 4; ++h) {
        float v = es[s * 4 + h] + ed[d * 4 + h];
        v = v > 0.f ? v : 0.2f * v;                      // leaky_relu(.,0.2)
        atomicMaxF(&mseg[d * 4 + h], v);
    }
}

__global__ void k_edge1_exp(const int* __restrict__ ei, const float* __restrict__ es,
                            const float* __restrict__ ed, const float* __restrict__ mseg,
                            float* __restrict__ exbuf, float* __restrict__ denom,
                            int E, int ET) {
    int e = blockIdx.x * blockDim.x + threadIdx.x;
    if (e >= ET) return;
    int s, d;
    if (e < E) { s = ei[e]; d = ei[E + e]; } else { s = d = e - E; }
#pragma unroll
    for (int h = 0; h < 4; ++h) {
        float v = es[s * 4 + h] + ed[d * 4 + h];
        v = v > 0.f ? v : 0.2f * v;
        float ex = __expf(v - mseg[d * 4 + h]);
        exbuf[e * 4 + h] = ex;
        atomicAdd(&denom[d * 4 + h], ex);
    }
}

__global__ void k_edge1_acc(const int* __restrict__ ei, const float* __restrict__ exbuf,
                            const float* __restrict__ denom, const float* __restrict__ h1,
                            float* __restrict__ accum, int E) {
    int e = blockIdx.x;
    int j = threadIdx.x;                               // 128
    int h = j >> 5;
    int s, d;
    if (e < E) { s = ei[e]; d = ei[E + e]; } else { s = d = e - E; }
    float alpha = exbuf[e * 4 + h] / (denom[d * 4 + h] + 1e-16f);
    atomicAdd(&accum[d * 128 + j], alpha * h1[s * 128 + j]);
}

__global__ void k_x1_final(const float* __restrict__ accum, const float* __restrict__ b1,
                           float* __restrict__ x1, unsigned short* __restrict__ x1bf) {
    int i = blockIdx.x * blockDim.x + threadIdx.x;
    if (i >= NN * 128) return;
    int j = i & 127;
    float v = accum[i] + b1[j];
    v = v > 0.f ? v : 0.f;
    x1[i] = v;
    x1bf[i] = f2bf(v);
}

// ---------------------------------------------------------------- markers
__global__ void k_colsum(const float* __restrict__ x1, float* __restrict__ stats) {
    int n = blockIdx.x * blockDim.x + threadIdx.x;
    if (n >= NN) return;
#pragma unroll
    for (int c = 0; c < 6; ++c) atomicAdd(&stats[c], x1[n * 128 + c]);
}

__global__ void k_nm(const float* __restrict__ x1, const float* __restrict__ stats,
                     unsigned int* __restrict__ nmbits, float* __restrict__ cnt) {
    int n = blockIdx.x * blockDim.x + threadIdx.x;
    if (n >= NN) return;
    unsigned int bits = 0;
#pragma unroll
    for (int c = 0; c < 6; ++c) {
        int b = x1[n * 128 + c] > stats[c] * (1.f / (float)NN);
        bits |= ((unsigned)b) << c;
        if (b) atomicAdd(&cnt[c], 1.f);
    }
    nmbits[n] = bits;
}

__global__ void k_maskprep(const unsigned int* __restrict__ nmbits,
                           const float* __restrict__ cnt, unsigned int* __restrict__ maskb,
                           float* __restrict__ rowinv) {
    int n = blockIdx.x * blockDim.x + threadIdx.x;
    if (n >= NN) return;
    const int m0[5] = {0, 2, 2, 1, 4};
    const int m1[5] = {3, 1, 5, 0, -1};
    unsigned int nb = nmbits[n];
#pragma unroll
    for (int k = 0; k < 5; ++k) {
        unsigned int bits = 0;
        float rs = 0.f;
        int a = m0[k];
        int ba = (nb >> a) & 1;
        bits |= (unsigned)ba;
        if (ba) rs += cnt[a];
        int b = m1[k];
        if (b >= 0) {
            int bb = (nb >> b) & 1;
            bits |= ((unsigned)bb) << 1;
            if (bb) rs += cnt[b];
        }
        maskb[k * NN + n] = bits;
        rowinv[k * NN + n] = 1.f / (rs + 1e-8f);
    }
}

// ---------------------------------------------------------------- WMMA NT GEMM
// C[m,n] = sum_k A[m,k]*B[n,k] + bias[n];   A:[M,K] bf16 row-major, B:[N,K] bf16 row-major
__global__ __launch_bounds__(128) void k_gemm_nt_bf16(
    const unsigned short* __restrict__ A, const unsigned short* __restrict__ B,
    const float* __restrict__ bias, unsigned short* __restrict__ C,
    int M, int Ncols, int K) {
    const int wid = blockIdx.x * 4 + (threadIdx.x >> 5);
    const int ntiles = Ncols >> 4;
    if (wid >= (M >> 4) * ntiles) return;
    const int mt = wid / ntiles, nt = wid % ntiles;
    const int lane = threadIdx.x & 31;
    const int hl = lane & 15, hs = lane >> 4;
    const unsigned short* arow = A + (size_t)(mt * 16 + hl) * K;
    const unsigned short* brow = B + (size_t)(nt * 16 + hl) * K;
    f32x8 acc = {0.f, 0.f, 0.f, 0.f, 0.f, 0.f, 0.f, 0.f};
    for (int ks = 0; ks < K; ks += 32) {
        bf16x16 a = load_a_frag(arow, ks + hs * 16);
        bf16x16 b = *reinterpret_cast<const bf16x16*>(brow + ks + hs * 16);
        acc = wmma_bf16(a, b, acc);
    }
    float bv = bias ? bias[nt * 16 + hl] : 0.f;
#pragma unroll
    for (int r = 0; r < 8; ++r) {
        int row = mt * 16 + hs * 8 + r;
        C[(size_t)row * Ncols + nt * 16 + hl] = f2bf(acc[r] + bv);
    }
}

// outsum[m,n] += 0.2 * (sum_k A[m,k]*B[n,k] + bias[n])
__global__ __launch_bounds__(128) void k_gemm_nt_accum(
    const unsigned short* __restrict__ A, const unsigned short* __restrict__ B,
    const float* __restrict__ bias, float* __restrict__ C,
    int M, int Ncols, int K) {
    const int wid = blockIdx.x * 4 + (threadIdx.x >> 5);
    const int ntiles = Ncols >> 4;
    if (wid >= (M >> 4) * ntiles) return;
    const int mt = wid / ntiles, nt = wid % ntiles;
    const int lane = threadIdx.x & 31;
    const int hl = lane & 15, hs = lane >> 4;
    const unsigned short* arow = A + (size_t)(mt * 16 + hl) * K;
    const unsigned short* brow = B + (size_t)(nt * 16 + hl) * K;
    f32x8 acc = {0.f, 0.f, 0.f, 0.f, 0.f, 0.f, 0.f, 0.f};
    for (int ks = 0; ks < K; ks += 32) {
        bf16x16 a = load_a_frag(arow, ks + hs * 16);
        bf16x16 b = *reinterpret_cast<const bf16x16*>(brow + ks + hs * 16);
        acc = wmma_bf16(a, b, acc);
    }
    float bv = bias ? bias[nt * 16 + hl] : 0.f;
#pragma unroll
    for (int r = 0; r < 8; ++r) {
        int row = mt * 16 + hs * 8 + r;
        C[(size_t)row * Ncols + nt * 16 + hl] += 0.2f * (acc[r] + bv);
    }
}

// ---------------------------------------------------------------- flash attention
// One wave owns 16 query rows; 4 waves/block; loop over 32-key tiles with online softmax.
// V tile staged into LDS via the Tensor Data Mover (2D tile descriptor), then
// transposed in-LDS so P@V B-fragments are single 32B ds loads.
__global__ __launch_bounds__(128) void k_attn(
    const unsigned short* __restrict__ qkv,          // [N][384] bf16: q|k|v
    const unsigned int* __restrict__ maskb,          // [N] packed marker bits
    const float* __restrict__ rowinv,                // [N] mask row normalizer
    unsigned short* __restrict__ obf) {              // [N][128] bf16 out
    __shared__ __align__(32) unsigned short Vr[32 * EMBD];   // [key][feat] (TDM dest)
    __shared__ __align__(32) unsigned short Vt[EMBD * 32];   // [feat][key]
    __shared__ __align__(32) unsigned short Pt[4][16 * 32];  // per-wave P tile
    const int wave = threadIdx.x >> 5;
    const int lane = threadIdx.x & 31;
    const int hl = lane & 15, hs = lane >> 4;
    const int r0 = blockIdx.x * 64 + wave * 16;
    const float scale = 0.08838834764831845f;        // 1/sqrt(128)

    // Q fragments (rows r0..r0+15, full K=128)
    bf16x16 qf[4];
    const unsigned short* qrow = qkv + (size_t)(r0 + hl) * QKVD;
#pragma unroll
    for (int s = 0; s < 4; ++s) qf[s] = load_a_frag(qrow, s * 32 + hs * 16);

    unsigned int rbits[8];
    float rinv[8];
    const int rbase = r0 + hs * 8;
#pragma unroll
    for (int r = 0; r < 8; ++r) {
        rbits[r] = maskb[rbase + r];
        rinv[r] = rowinv[rbase + r];
    }

    float mrow[8], lrow[8];
    f32x8 Of[8];
#pragma unroll
    for (int r = 0; r < 8; ++r) { mrow[r] = -3.0e38f; lrow[r] = 0.f; }
#pragma unroll
    for (int ct = 0; ct < 8; ++ct)
#pragma unroll
        for (int r = 0; r < 8; ++r) Of[ct][r] = 0.f;

#if USE_TDM
    // LDS aperture: low 32 bits of the flat shared address = LDS byte offset
    const unsigned vr_lds = (unsigned)(uintptr_t)(void*)&Vr[0];
#endif

    for (int j0 = 0; j0 < NN; j0 += 32) {
        __syncthreads();   // previous tile fully consumed before restaging

#if USE_TDM
        if (wave == 0) {
            // Tensor DMA: 2D tile [32 keys][128 feats] bf16 from qkv(+v offset) -> Vr
            unsigned long long ga =
                (unsigned long long)(uintptr_t)(qkv + (size_t)j0 * QKVD + 256);
            u32x4 g0 = {1u,                               // count=1, user mode
                        vr_lds,                           // lds_addr (bytes)
                        (unsigned)(ga & 0xFFFFFFFFull),   // global_addr[31:0]
                        (unsigned)((ga >> 32) & 0x1FFFFFFull) | 0x80000000u}; // type=2
            i32x8 g1 = {(int)0x00010000,       // data_size=1 (2 bytes), wg_mask=0
                        (int)(128u << 16),     // tensor_dim0 = 128 (bits 48..63)
                        (int)0x10000000,       // tensor_dim1 = 4096 (bits 80..95)
                        (int)(128u << 16),     // tile_dim0 = 128 (bits 112..127)
                        32,                    // tile_dim1 = 32
                        384,                   // tensor_dim0_stride = 384
                        0, 0};
            i32x4 z4 = {0, 0, 0, 0};
#if __clang_major__ >= 23
            i32x8 z8 = {0, 0, 0, 0, 0, 0, 0, 0};
            __builtin_amdgcn_tensor_load_to_lds(g0, g1, z4, z4, z8, 0);
#else
            __builtin_amdgcn_tensor_load_to_lds(g0, g1, z4, z4, 0);
#endif
        }
        __builtin_amdgcn_s_wait_tensorcnt(0);
        __syncthreads();
        {
            // in-LDS transpose Vr[key][feat] -> Vt[feat][key]
            int key = threadIdx.x >> 2;
            int f0 = (threadIdx.x & 3) * 32;
#pragma unroll
            for (int i = 0; i < 32; ++i) Vt[(f0 + i) * 32 + key] = Vr[key * 128 + f0 + i];
        }
#else
        {
            // fallback: gather V^T straight from global
            int key = threadIdx.x >> 2;
            int f0 = (threadIdx.x & 3) * 32;
            const unsigned short* vrow = qkv + (size_t)(j0 + key) * QKVD + 256 + f0;
#pragma unroll
            for (int i = 0; i < 32; ++i) Vt[(f0 + i) * 32 + key] = vrow[i];
        }
#endif
        __syncthreads();

        // S = Q * K^T for two 16-key halves
        f32x8 S0 = {0.f, 0.f, 0.f, 0.f, 0.f, 0.f, 0.f, 0.f};
        f32x8 S1 = {0.f, 0.f, 0.f, 0.f, 0.f, 0.f, 0.f, 0.f};
        const unsigned short* krow0 = qkv + (size_t)(j0 + hl) * QKVD + 128;
        const unsigned short* krow1 = qkv + (size_t)(j0 + 16 + hl) * QKVD + 128;
#pragma unroll
        for (int s = 0; s < 4; ++s) {
            bf16x16 b0 = *reinterpret_cast<const bf16x16*>(krow0 + s * 32 + hs * 16);
            bf16x16 b1 = *reinterpret_cast<const bf16x16*>(krow1 + s * 32 + hs * 16);
            S0 = wmma_bf16(qf[s], b0, S0);
            S1 = wmma_bf16(qf[s], b1, S1);
        }

        unsigned int cb0 = maskb[j0 + hl], cb1 = maskb[j0 + 16 + hl];
        float p0[8], p1[8];
#pragma unroll
        for (int r = 0; r < 8; ++r) {
            float e0 = S0[r] * scale + rinv[r] * (float)__popc(rbits[r] & cb0);
            float e1 = S1[r] * scale + rinv[r] * (float)__popc(rbits[r] & cb1);
            float t = fmaxf(e0, e1);
#pragma unroll
            for (int o = 8; o >= 1; o >>= 1) t = fmaxf(t, __shfl_xor(t, o, 16));
            float mn = fmaxf(mrow[r], t);
            float corr = __expf(mrow[r] - mn);
            e0 = __expf(e0 - mn);
            e1 = __expf(e1 - mn);
            float rs = e0 + e1;
#pragma unroll
            for (int o = 8; o >= 1; o >>= 1) rs += __shfl_xor(rs, o, 16);
            lrow[r] = lrow[r] * corr + rs;
            mrow[r] = mn;
            p0[r] = e0;
            p1[r] = e1;
#pragma unroll
            for (int ct = 0; ct < 8; ++ct) Of[ct][r] *= corr;
        }

        // write P tile (16 rows x 32 keys) to this wave's LDS region
        unsigned short* pw = Pt[wave];
#pragma unroll
        for (int r = 0; r < 8; ++r) {
            pw[(hs * 8 + r) * 32 + hl] = f2bf(p0[r]);
            pw[(hs * 8 + r) * 32 + 16 + hl] = f2bf(p1[r]);
        }
        __syncthreads();

        // O += P @ V   (A = P 16x32, B = V^T slices 32x16)
        bf16x16 aP = load_a_frag(pw + hl * 32, hs * 16);
#pragma unroll
        for (int ct = 0; ct < 8; ++ct) {
            bf16x16 bV =
                *reinterpret_cast<const bf16x16*>(&Vt[(ct * 16 + hl) * 32 + hs * 16]);
            Of[ct] = wmma_bf16(aP, bV, Of[ct]);
        }
    }

    // normalize and store bf16
#pragma unroll
    for (int r = 0; r < 8; ++r) {
        float inv = 1.f / (lrow[r] + 1e-16f);
#pragma unroll
        for (int ct = 0; ct < 8; ++ct)
            obf[(size_t)(rbase + r) * EMBD + ct * 16 + hl] = f2bf(Of[ct][r] * inv);
    }
}

// ---------------------------------------------------------------- tail
__global__ void k_x2(const float* __restrict__ x1, const float* __restrict__ outsum,
                     float* __restrict__ x2) {
    int i = blockIdx.x * blockDim.x + threadIdx.x;
    if (i < NN * 128) x2[i] = x1[i] + outsum[i];
}

__global__ void k_gat2_node(const float* __restrict__ x2, const float* __restrict__ W2,
                            float* __restrict__ h2) {
    int n = blockIdx.x, j = threadIdx.x;     // 64 threads
    float acc = 0.f;
    for (int d = 0; d < 128; ++d) acc += x2[n * 128 + d] * W2[d * 64 + j];
    h2[n * 64 + j] = acc;
}

__global__ void k_gat2_coef(const float* __restrict__ h2, const float* __restrict__ as,
                            const float* __restrict__ ad, float* __restrict__ es,
                            float* __restrict__ ed) {
    int n = blockIdx.x * blockDim.x + threadIdx.x;
    if (n >= NN) return;
    float s = 0.f, d = 0.f;
#pragma unroll
    for (int f = 0; f < 64; ++f) {
        float v = h2[n * 64 + f];
        s += v * as[f];
        d += v * ad[f];
    }
    es[n] = s;
    ed[n] = d;
}

__global__ void k_edge2_max(const int* __restrict__ ei, const float* __restrict__ es,
                            const float* __restrict__ ed, float* __restrict__ mseg,
                            int E, int ET) {
    int e = blockIdx.x * blockDim.x + threadIdx.x;
    if (e >= ET) return;
    int s, d;
    if (e < E) { s = ei[e]; d = ei[E + e]; } else { s = d = e - E; }
    float v = es[s] + ed[d];
    v = v > 0.f ? v : 0.2f * v;
    atomicMaxF(&mseg[d], v);
}

__global__ void k_edge2_exp(const int* __restrict__ ei, const float* __restrict__ es,
                            const float* __restrict__ ed, const float* __restrict__ mseg,
                            float* __restrict__ exbuf, float* __restrict__ denom,
                            int E, int ET) {
    int e = blockIdx.x * blockDim.x + threadIdx.x;
    if (e >= ET) return;
    int s, d;
    if (e < E) { s = ei[e]; d = ei[E + e]; } else { s = d = e - E; }
    float v = es[s] + ed[d];
    v = v > 0.f ? v : 0.2f * v;
    float ex = __expf(v - mseg[d]);
    exbuf[e] = ex;
    atomicAdd(&denom[d], ex);
}

__global__ void k_edge2_acc(const int* __restrict__ ei, const float* __restrict__ exbuf,
                            const float* __restrict__ denom, const float* __restrict__ h2,
                            float* __restrict__ accum, int E) {
    int e = blockIdx.x;
    int j = threadIdx.x;                     // 64
    int s, d;
    if (e < E) { s = ei[e]; d = ei[E + e]; } else { s = d = e - E; }
    float alpha = exbuf[e] / (denom[d] + 1e-16f);
    atomicAdd(&accum[d * 64 + j], alpha * h2[s * 64 + j]);
}

__global__ void k_x3_final(const float* __restrict__ accum, const float* __restrict__ b2,
                           float* __restrict__ x3) {
    int i = blockIdx.x * blockDim.x + threadIdx.x;
    if (i >= NN * 64) return;
    float v = accum[i] + b2[i & 63];
    x3[i] = v > 0.f ? v : 0.f;
}

__global__ void k_final(const float* __restrict__ x3, const float* __restrict__ fW,
                        const float* __restrict__ fb, float* __restrict__ out) {
    int n = blockIdx.x * blockDim.x + threadIdx.x;
    if (n >= NN) return;
#pragma unroll
    for (int c = 0; c < 6; ++c) {
        float a = fb[c];
        for (int j = 0; j < 64; ++j) a += x3[n * 64 + j] * fW[j * 6 + c];
        out[n * 6 + c] = a;
    }
}

// ---------------------------------------------------------------- launch
extern "C" void kernel_launch(void* const* d_in, const int* in_sizes, int n_in,
                              void* d_out, int out_size, void* d_ws, size_t ws_size,
                              hipStream_t stream) {
    const float* x    = (const float*)d_in[0];
    const int*   ei   = (const int*)d_in[1];
    const float* W1   = (const float*)d_in[3];
    const float* a1s  = (const float*)d_in[4];
    const float* a1d  = (const float*)d_in[5];
    const float* b1   = (const float*)d_in[6];
    const float* W2   = (const float*)d_in[7];
    const float* a2s  = (const float*)d_in[8];
    const float* a2d  = (const float*)d_in[9];
    const float* b2   = (const float*)d_in[10];
    const float* ipw  = (const float*)d_in[11];
    const float* ipb  = (const float*)d_in[12];
    const float* ow   = (const float*)d_in[13];
    const float* ob   = (const float*)d_in[14];
    const float* fW   = (const float*)d_in[15];
    const float* fb   = (const float*)d_in[16];
    float* out = (float*)d_out;

    const int E  = in_sizes[1] / 2;        // 65536
    const int ET = E + NN;                 // + self loops

    // workspace carve-up (all offsets 64-float = 256 B aligned)
    float* base = (float*)d_ws;
    size_t off = 0;
    auto allocF = [&](size_t n) -> float* {
        float* p = base + off;
        off += (n + 63) & ~(size_t)63;
        return p;
    };
    auto allocU = [&](size_t n) -> unsigned short* {
        return (unsigned short*)allocF((n + 1) / 2);
    };

    float* h1     = allocF((size_t)NN * 128);
    float* es     = allocF((size_t)NN * 4);
    float* ed     = allocF((size_t)NN * 4);
    float* mseg   = allocF((size_t)NN * 4);
    float* denom  = allocF((size_t)NN * 4);
    float* exbuf  = allocF((size_t)ET * 4);
    float* accum  = allocF((size_t)NN * 128);
    float* x1     = allocF((size_t)NN * 128);
    unsigned short* x1bf = allocU((size_t)NN * 128);
    float* stats  = allocF(8);
    float* cnt    = allocF(8);
    unsigned int* nmbits = (unsigned int*)allocF(NN);
    unsigned int* maskbA = (unsigned int*)allocF((size_t)5 * NN);
    float* rowinvA = allocF((size_t)5 * NN);
    unsigned short* wbf   = allocU((size_t)5 * 384 * 128);
    unsigned short* owbf  = allocU((size_t)5 * 128 * 128);
    unsigned short* qkvbf = allocU((size_t)NN * QKVD);
    unsigned short* obf   = allocU((size_t)NN * 128);
    float* outsum = allocF((size_t)NN * 128);
    float* x2     = allocF((size_t)NN * 128);
    float* h2     = allocF((size_t)NN * 64);
    float* es2    = allocF(NN);
    float* ed2    = allocF(NN);
    float* mseg2  = allocF(NN);
    float* denom2 = allocF(NN);
    float* exbuf2 = allocF(ET);
    float* accum2 = allocF((size_t)NN * 64);
    float* x3     = allocF((size_t)NN * 64);
    (void)ws_size; (void)n_in; (void)out_size;

    auto G = [](int n, int b) { return (n + b - 1) / b; };

    // init
    k_fill<<<G(NN * 4, 256), 256, 0, stream>>>(mseg, -3.0e38f, NN * 4);
    k_fill<<<G(NN * 4, 256), 256, 0, stream>>>(denom, 0.f, NN * 4);
    k_fill<<<G(NN * 128, 256), 256, 0, stream>>>(accum, 0.f, NN * 128);
    k_fill<<<G(NN * 128, 256), 256, 0, stream>>>(outsum, 0.f, NN * 128);
    k_fill<<<1, 64, 0, stream>>>(stats, 0.f, 8);
    k_fill<<<1, 64, 0, stream>>>(cnt, 0.f, 8);
    k_fill<<<G(NN, 256), 256, 0, stream>>>(mseg2, -3.0e38f, NN);
    k_fill<<<G(NN, 256), 256, 0, stream>>>(denom2, 0.f, NN);
    k_fill<<<G(NN * 64, 256), 256, 0, stream>>>(accum2, 0.f, NN * 64);

    // bf16 weights
    k_f2bf<<<G(5 * 384 * 128, 256), 256, 0, stream>>>(ipw, wbf, 5 * 384 * 128);
    k_f2bf<<<G(5 * 128 * 128, 256), 256, 0, stream>>>(ow, owbf, 5 * 128 * 128);

    // GAT layer 1
    k_gat1_node<<<NN, 128, 0, stream>>>(x, W1, h1);
    k_gat1_coef<<<G(NN * 4, 256), 256, 0, stream>>>(h1, a1s, a1d, es, ed);
    k_edge1_max<<<G(ET, 256), 256, 0, stream>>>(ei, es, ed, mseg, E, ET);
    k_edge1_exp<<<G(ET, 256), 256, 0, stream>>>(ei, es, ed, mseg, exbuf, denom, E, ET);
    k_edge1_acc<<<ET, 128, 0, stream>>>(ei, exbuf, denom, h1, accum, E);
    k_x1_final<<<G(NN * 128, 256), 256, 0, stream>>>(accum, b1, x1, x1bf);

    // marker masks
    k_colsum<<<G(NN, 256), 256, 0, stream>>>(x1, stats);
    k_nm<<<G(NN, 256), 256, 0, stream>>>(x1, stats, nmbits, cnt);
    k_maskprep<<<G(NN, 256), 256, 0, stream>>>(nmbits, cnt, maskbA, rowinvA);

    // 5 attention blocks (WMMA heavy path)
    for (int k = 0; k < 5; ++k) {
        int qkv_tiles = (NN / 16) * (QKVD / 16);
        k_gemm_nt_bf16<<<G(qkv_tiles, 4), 128, 0, stream>>>(
            x1bf, wbf + (size_t)k * 384 * 128, ipb + (size_t)k * 384, qkvbf,
            NN, QKVD, 128);
        k_attn<<<NN / 64, 128, 0, stream>>>(qkvbf, maskbA + (size_t)k * NN,
                                            rowinvA + (size_t)k * NN, obf);
        int op_tiles = (NN / 16) * (128 / 16);
        k_gemm_nt_accum<<<G(op_tiles, 4), 128, 0, stream>>>(
            obf, owbf + (size_t)k * 128 * 128, ob + (size_t)k * 128, outsum,
            NN, 128, 128);
    }

    k_x2<<<G(NN * 128, 256), 256, 0, stream>>>(x1, outsum, x2);

    // GAT layer 2
    k_gat2_node<<<NN, 64, 0, stream>>>(x2, W2, h2);
    k_gat2_coef<<<G(NN, 256), 256, 0, stream>>>(h2, a2s, a2d, es2, ed2);
    k_edge2_max<<<G(ET, 256), 256, 0, stream>>>(ei, es2, ed2, mseg2, E, ET);
    k_edge2_exp<<<G(ET, 256), 256, 0, stream>>>(ei, es2, ed2, mseg2, exbuf2, denom2, E, ET);
    k_edge2_acc<<<ET, 64, 0, stream>>>(ei, exbuf2, denom2, h2, accum2, E);
    k_x3_final<<<G(NN * 64, 256), 256, 0, stream>>>(accum2, b2, x3);

    // final projection
    k_final<<<G(NN, 256), 256, 0, stream>>>(x3, fW, fb, out);
}
